// GFCNA_69939247448924
// MI455X (gfx1250) — matrix operand order repaired and checked by reading, not compiled, not run on hardware
//
#include <hip/hip_runtime.h>

// ---------------------------------------------------------------------------
// SplineConv U-Net forward for MI455X (gfx1250, wave32, WMMA).
//
//   1. Node-level batched GEMMs Z[k] = X @ W[k] (k = 0..K*K-1, root as slice
//      K*K) via v_wmma_f32_16x16x32_f16.  Shapes are template parameters so
//      fragment fills are branch-free vector loads (EXEC all-ones at WMMA).
//      in=1 (outer product) and out=1 (matvec) cases use scalar kernels.
//   2. Edge stage is memory-bound: gather 4 Z rows per edge weighted by the
//      degree-1 B-spline basis, atomicAdd into agg (32 MB, L2-resident).
//   3. finalize: agg/deg + Zroot + bias, ELU.
//   4. Pooling: segment_max via monotone float<->int map + atomicMax(int);
//      segment means + Cartesian edge attrs via atomic reductions.
// ---------------------------------------------------------------------------

#define N0c 262144
#define E0c 2097152
#define N1c 131072
#define E1c 1048576
#define N2c 65536
#define E2c 524288

typedef __attribute__((ext_vector_type(16))) _Float16 v16h;
typedef __attribute__((ext_vector_type(8)))  float    v8f;

__device__ __forceinline__ int f32_ord(float f) {
  int i = __float_as_int(f);
  return (i >= 0) ? i : (i ^ 0x7FFFFFFF);   // monotone, involutive on sign branch
}
__device__ __forceinline__ float ord_f32(int m) {
  int b = (m >= 0) ? m : (m ^ 0x7FFFFFFF);
  return __int_as_float(b);
}

__global__ void fill_i32_kernel(int* __restrict__ p, int v, int n) {
  int i = blockIdx.x * blockDim.x + threadIdx.x;
  if (i < n) p[i] = v;
}

// --------------------------- GEMM kernels ----------------------------------

// Z[k][node][oc] = sum_kk X[node][kk] * Wk[kk][oc]; slice k==nkmain uses root.
// One wave computes a 16(M) x 16(N) tile; 8 waves/block -> 128 rows per block.
// nnodes must be a multiple of 128; IN_C in {32,64}; OUT_C in {32,64}.
template <int IN_C, int OUT_C>
__global__ __launch_bounds__(256) void gemm_wmma_kernel(
    const float* __restrict__ X, const float* __restrict__ W,
    const float* __restrict__ root, float* __restrict__ Z,
    int nnodes, int nkmain) {
  const int lane   = threadIdx.x & 31;
  const int wave   = threadIdx.x >> 5;
  const int laneHi = lane >> 4;      // 0: lanes 0-15, 1: lanes 16-31
  const int lmod   = lane & 15;
  const int mbase  = (blockIdx.x * 8 + wave) * 16;
  const int col    = blockIdx.y * 16 + lmod;
  const int k      = blockIdx.z;
  const float* Wk  = (k < nkmain) ? (W + (size_t)k * IN_C * OUT_C) : root;

  v8f c = {0.f, 0.f, 0.f, 0.f, 0.f, 0.f, 0.f, 0.f};
  const int mrow = mbase + lmod;
#pragma unroll
  for (int kb = 0; kb < IN_C / 32; ++kb) {
    // A fragment (16x32 f16): lane row = mrow; halves 0-7 -> K = kbase+0..7,
    // halves 8-15 -> K = kbase+16..23, kbase = kb*32 + 8*laneHi.
    const int kbase = kb * 32 + laneHi * 8;    // multiple of 8 -> 16B aligned
    const float4* rp =
        reinterpret_cast<const float4*>(X + (size_t)mrow * IN_C + kbase);
    float4 x0 = rp[0], x1 = rp[1], x2 = rp[4], x3 = rp[5];
    v16h a;
    a[0]  = (_Float16)x0.x; a[1]  = (_Float16)x0.y;
    a[2]  = (_Float16)x0.z; a[3]  = (_Float16)x0.w;
    a[4]  = (_Float16)x1.x; a[5]  = (_Float16)x1.y;
    a[6]  = (_Float16)x1.z; a[7]  = (_Float16)x1.w;
    a[8]  = (_Float16)x2.x; a[9]  = (_Float16)x2.y;
    a[10] = (_Float16)x2.z; a[11] = (_Float16)x2.w;
    a[12] = (_Float16)x3.x; a[13] = (_Float16)x3.y;
    a[14] = (_Float16)x3.z; a[15] = (_Float16)x3.w;
    // B fragment (32x16 f16): half h = B[K = kb*32+8*laneHi+(h<8?h:h+8)][N = col].
    v16h b;
#pragma unroll
    for (int h = 0; h < 16; ++h) {
      int kk = kb * 32 + laneHi * 8 + (h < 8 ? h : h + 8);
      b[h] = (_Float16)Wk[kk * OUT_C + col];
    }
    c = __builtin_amdgcn_wmma_f32_16x16x32_f16(false, a, false, b, (short)0, c,
                                               false, false);
  }
  float* Zk = Z + (size_t)k * nnodes * OUT_C;
#pragma unroll
  for (int r = 0; r < 8; ++r) {
    int row = mbase + r + 8 * laneHi;   // C/D layout: VGPR r -> M = r + 8*laneHi
    Zk[(size_t)row * OUT_C + col] = c[r];
  }
}

// in_c == 1: pure outer product Z[k][n][oc] = x[n] * Wk[oc].
__global__ void gemm_in1_kernel(const float* __restrict__ X,
                                const float* __restrict__ W,
                                const float* __restrict__ root,
                                float* __restrict__ Z,
                                int nnodes, int out_c, int nkmain) {
  int idx = blockIdx.x * blockDim.x + threadIdx.x;
  int per = nnodes * out_c;
  if (idx >= (nkmain + 1) * per) return;
  int k = idx / per;
  int r = idx - k * per;
  int n = r / out_c;
  int oc = r - n * out_c;
  const float* Wk = (k < nkmain) ? (W + (size_t)k * out_c) : root;
  Z[idx] = X[n] * Wk[oc];
}

// out_c == 1: matvec Z[k][n] = dot(X[n,:], Wk[:,0]).
__global__ void gemm_out1_kernel(const float* __restrict__ X,
                                 const float* __restrict__ W,
                                 const float* __restrict__ root,
                                 float* __restrict__ Z,
                                 int nnodes, int in_c, int nkmain) {
  int idx = blockIdx.x * blockDim.x + threadIdx.x;
  if (idx >= (nkmain + 1) * nnodes) return;
  int k = idx / nnodes;
  int n = idx - k * nnodes;
  const float* Wk = (k < nkmain) ? (W + (size_t)k * in_c) : root;
  const float* xr = X + (size_t)n * in_c;
  float acc = 0.f;
  for (int i = 0; i < in_c; ++i) acc += xr[i] * Wk[i];
  Z[idx] = acc;
}

// --------------------------- Edge / node kernels ---------------------------

__global__ void deg_kernel(const int* __restrict__ ei, int E, float* __restrict__ deg) {
  int e = blockIdx.x * blockDim.x + threadIdx.x;
  if (e < E) atomicAdd(&deg[ei[E + e]], 1.0f);
}

// Per (edge, oc): evaluate the 4-tap bilinear spline basis, gather 4 Z rows,
// scatter-add into agg[dst].
template <int OUT_C, int K>
__global__ void edge_accum_kernel(const float* __restrict__ Z,
                                  const float* __restrict__ ea,
                                  const int* __restrict__ ei,
                                  float* __restrict__ agg,
                                  int E, int nnodes) {
  int idx = blockIdx.x * blockDim.x + threadIdx.x;
  if (idx >= E * OUT_C) return;
  int e = idx / OUT_C;
  int oc = idx - e * OUT_C;
  float p0 = ea[2 * e], p1 = ea[2 * e + 1];
  float v0 = fminf(fmaxf(p0, 0.f), 1.f) * (float)(K - 1);
  float v1 = fminf(fmaxf(p1, 0.f), 1.f) * (float)(K - 1);
  int lo0 = (int)v0; if (lo0 > K - 2) lo0 = K - 2;
  int lo1 = (int)v1; if (lo1 > K - 2) lo1 = K - 2;
  float f0 = v0 - (float)lo0, f1 = v1 - (float)lo1;
  int src = ei[e], dst = ei[E + e];
  const float* zs = Z + (size_t)src * OUT_C + oc;
  size_t ks = (size_t)nnodes * OUT_C;
  float m = (1.f - f0) * (1.f - f1) * zs[(size_t)(lo0     + lo1       * K) * ks]
          +        f0  * (1.f - f1) * zs[(size_t)(lo0 + 1 + lo1       * K) * ks]
          + (1.f - f0) *        f1  * zs[(size_t)(lo0     + (lo1 + 1) * K) * ks]
          +        f0  *        f1  * zs[(size_t)(lo0 + 1 + (lo1 + 1) * K) * ks];
  atomicAdd(&agg[(size_t)dst * OUT_C + oc], m);
}

__global__ void finalize_kernel(const float* __restrict__ agg,
                                const float* __restrict__ deg,
                                const float* __restrict__ Zroot,
                                const float* __restrict__ bias,
                                float* __restrict__ outp, int nnodes, int out_c) {
  int idx = blockIdx.x * blockDim.x + threadIdx.x;
  if (idx >= nnodes * out_c) return;
  int n = idx / out_c;
  int oc = idx - n * out_c;
  float v = agg[idx] / fmaxf(deg[n], 1.f) + Zroot[idx] + bias[oc];
  outp[idx] = (v > 0.f) ? v : (expf(v) - 1.f);   // ELU(alpha=1)
}

__global__ void pool_max_kernel(const float* __restrict__ x,
                                const int* __restrict__ cluster,
                                int* __restrict__ aggI, int nsrc, int ch) {
  int idx = blockIdx.x * blockDim.x + threadIdx.x;
  if (idx >= nsrc * ch) return;
  int n = idx / ch;
  int oc = idx - n * ch;
  atomicMax(&aggI[(size_t)cluster[n] * ch + oc], f32_ord(x[idx]));
}

__global__ void unmap_kernel(const int* __restrict__ aggI, float* __restrict__ o, int n) {
  int i = blockIdx.x * blockDim.x + threadIdx.x;
  if (i < n) o[i] = ord_f32(aggI[i]);
}

__global__ void pos_accum_kernel(const float* __restrict__ pos,
                                 const int* __restrict__ cluster,
                                 float* __restrict__ psum, float* __restrict__ pcnt,
                                 int nsrc) {
  int n = blockIdx.x * blockDim.x + threadIdx.x;
  if (n >= nsrc) return;
  int c = cluster[n];
  atomicAdd(&psum[2 * c],     pos[2 * n]);
  atomicAdd(&psum[2 * c + 1], pos[2 * n + 1]);
  atomicAdd(&pcnt[c], 1.0f);
}

__global__ void pos_div_kernel(const float* __restrict__ psum,
                               const float* __restrict__ pcnt,
                               float* __restrict__ po, int nd) {
  int n = blockIdx.x * blockDim.x + threadIdx.x;
  if (n >= nd) return;
  float cinv = 1.f / fmaxf(pcnt[n], 1.f);
  po[2 * n]     = psum[2 * n] * cinv;
  po[2 * n + 1] = psum[2 * n + 1] * cinv;
}

__global__ void cart_max_kernel(const float* __restrict__ pos,
                                const int* __restrict__ ei, int E,
                                int* __restrict__ mx) {
  int e = blockIdx.x * blockDim.x + threadIdx.x;
  if (e >= E) return;
  int s = ei[e], d = ei[E + e];
  float dx = pos[2 * s] - pos[2 * d];
  float dy = pos[2 * s + 1] - pos[2 * d + 1];
  float m = fmaxf(fabsf(dx), fabsf(dy));
  atomicMax(mx, __float_as_int(m));   // m >= 0: int order == float order
}

__global__ void cart_ea_kernel(const float* __restrict__ pos,
                               const int* __restrict__ ei, int E,
                               const int* __restrict__ mx, float* __restrict__ ea) {
  int e = blockIdx.x * blockDim.x + threadIdx.x;
  if (e >= E) return;
  int s = ei[e], d = ei[E + e];
  float m = fmaxf(__int_as_float(*mx), 1e-12f);
  float inv = 1.f / (2.f * m);
  ea[2 * e]     = (pos[2 * s]     - pos[2 * d])     * inv + 0.5f;
  ea[2 * e + 1] = (pos[2 * s + 1] - pos[2 * d + 1]) * inv + 0.5f;
}

__global__ void add_up_kernel(const float* __restrict__ xfr,
                              const int* __restrict__ cluster2,
                              const float* __restrict__ sp1,
                              float* __restrict__ y1, int n1) {
  int m = blockIdx.x * blockDim.x + threadIdx.x;
  if (m < n1) y1[m] = xfr[cluster2[m]] + sp1[m];
}

__global__ void sig_out_kernel(const float* __restrict__ y1,
                               const int* __restrict__ cluster1,
                               float* __restrict__ o, int n0) {
  int n = blockIdx.x * blockDim.x + threadIdx.x;
  if (n >= n0) return;
  float v = y1[cluster1[n]];
  o[n] = 1.f / (1.f + expf(-v));
}

// ------------------------ Conv orchestration (host) ------------------------

static void run_conv(const float* X, const float* W, const float* root,
                     const float* bias, const float* ea, const int* ei, int E,
                     const float* deg, float* Z, float* agg, float* outp,
                     int nnodes, int in_c, int out_c, int K, hipStream_t s) {
  const int nk = K * K;
  // 1) Z-GEMMs.
  if (in_c == 1) {
    int tot = (nk + 1) * nnodes * out_c;
    gemm_in1_kernel<<<(tot + 255) / 256, 256, 0, s>>>(X, W, root, Z, nnodes,
                                                      out_c, nk);
  } else if (out_c == 1) {
    int tot = (nk + 1) * nnodes;
    gemm_out1_kernel<<<(tot + 255) / 256, 256, 0, s>>>(X, W, root, Z, nnodes,
                                                       in_c, nk);
  } else if (in_c == 32 && out_c == 32) {
    dim3 gg(nnodes / 128, 2, nk + 1);
    gemm_wmma_kernel<32, 32><<<gg, 256, 0, s>>>(X, W, root, Z, nnodes, nk);
  } else if (in_c == 32 && out_c == 64) {
    dim3 gg(nnodes / 128, 4, nk + 1);
    gemm_wmma_kernel<32, 64><<<gg, 256, 0, s>>>(X, W, root, Z, nnodes, nk);
  } else {  // 64 -> 64
    dim3 gg(nnodes / 128, 4, nk + 1);
    gemm_wmma_kernel<64, 64><<<gg, 256, 0, s>>>(X, W, root, Z, nnodes, nk);
  }
  // 2) Edge gather/scatter.
  hipMemsetAsync(agg, 0, (size_t)nnodes * out_c * sizeof(float), s);
  int tot = E * out_c;
  dim3 eg((tot + 255) / 256);
  if (out_c == 32) {
    edge_accum_kernel<32, 5><<<eg, 256, 0, s>>>(Z, ea, ei, agg, E, nnodes);
  } else if (out_c == 64) {
    edge_accum_kernel<64, 3><<<eg, 256, 0, s>>>(Z, ea, ei, agg, E, nnodes);
  } else {
    edge_accum_kernel<1, 3><<<eg, 256, 0, s>>>(Z, ea, ei, agg, E, nnodes);
  }
  // 3) finalize + ELU.
  const float* Zroot = Z + (size_t)nk * nnodes * out_c;
  int nt = nnodes * out_c;
  finalize_kernel<<<(nt + 255) / 256, 256, 0, s>>>(agg, deg, Zroot, bias, outp,
                                                   nnodes, out_c);
}

extern "C" void kernel_launch(void* const* d_in, const int* in_sizes, int n_in,
                              void* d_out, int out_size, void* d_ws, size_t ws_size,
                              hipStream_t stream) {
  (void)in_sizes; (void)n_in; (void)out_size; (void)ws_size;
  // Inputs in setup_inputs() dict order; params pytree in JAX alphabetical order.
  const float* x    = (const float*)d_in[0];
  const float* pos  = (const float*)d_in[1];
  const float* eat0 = (const float*)d_in[2];
  const int*   ei0  = (const int*)d_in[3];
  const int*   cl1  = (const int*)d_in[4];
  const int*   ei1  = (const int*)d_in[5];
  const int*   cl2  = (const int*)d_in[6];
  const int*   ei2  = (const int*)d_in[7];
  const float* b1a = (const float*)d_in[8];   const float* b1b = (const float*)d_in[9];
  const float* b2a = (const float*)d_in[10];  const float* b2b = (const float*)d_in[11];
  const float* b3a = (const float*)d_in[12];  const float* b3b = (const float*)d_in[13];
  const float* bfr = (const float*)d_in[14];  const float* bp1 = (const float*)d_in[15];
  const float* r1a = (const float*)d_in[16];  const float* r1b = (const float*)d_in[17];
  const float* r2a = (const float*)d_in[18];  const float* r2b = (const float*)d_in[19];
  const float* r3a = (const float*)d_in[20];  const float* r3b = (const float*)d_in[21];
  const float* rfr = (const float*)d_in[22];  const float* rp1 = (const float*)d_in[23];
  const float* w1a = (const float*)d_in[24];  const float* w1b = (const float*)d_in[25];
  const float* w2a = (const float*)d_in[26];  const float* w2b = (const float*)d_in[27];
  const float* w3a = (const float*)d_in[28];  const float* w3b = (const float*)d_in[29];
  const float* wfr = (const float*)d_in[30];  const float* wp1 = (const float*)d_in[31];

  // Workspace carve-up (~1.15 GB; dominated by the 26-slice level-0 Z table).
  char* wsb = (char*)d_ws;
  size_t off = 0;
  auto A = [&](size_t bytes) -> void* {
    void* p = wsb + off;
    off = (off + bytes + 255) & ~(size_t)255;
    return p;
  };
  float* zbuf   = (float*)A((size_t)26 * N0c * 32 * sizeof(float));
  float* agg    = (float*)A((size_t)N0c * 32 * sizeof(float)); // >= N1*64, N2*64
  float* deg0   = (float*)A((size_t)N0c * sizeof(float));
  float* deg1   = (float*)A((size_t)N1c * sizeof(float));
  float* deg2   = (float*)A((size_t)N2c * sizeof(float));
  float* h0a    = (float*)A((size_t)N0c * 32 * sizeof(float));
  float* h0b    = (float*)A((size_t)N0c * 32 * sizeof(float));
  float* x1pool = (float*)A((size_t)N1c * 32 * sizeof(float));
  float* x1a    = (float*)A((size_t)N1c * 64 * sizeof(float));
  float* x1b    = (float*)A((size_t)N1c * 64 * sizeof(float));
  float* x2pool = (float*)A((size_t)N2c * 64 * sizeof(float));
  float* x2a    = (float*)A((size_t)N2c * 64 * sizeof(float));
  float* x2b    = (float*)A((size_t)N2c * 64 * sizeof(float));
  float* xfr    = (float*)A((size_t)N2c * sizeof(float));
  float* sp1    = (float*)A((size_t)N1c * sizeof(float));
  float* y1     = (float*)A((size_t)N1c * sizeof(float));
  float* pos1   = (float*)A((size_t)N1c * 2 * sizeof(float));
  float* pos2   = (float*)A((size_t)N2c * 2 * sizeof(float));
  float* psum   = (float*)A((size_t)N1c * 2 * sizeof(float));
  float* pcnt   = (float*)A((size_t)N1c * sizeof(float));
  float* ea1    = (float*)A((size_t)E1c * 2 * sizeof(float));
  float* ea2    = (float*)A((size_t)E2c * 2 * sizeof(float));
  int*   maxI   = (int*)A(256);
  int*   aggI   = (int*)agg;   // pooling reuses the agg region

  const int B = 256;
  // Degrees (shared per level across convs).
  hipMemsetAsync(deg0, 0, (size_t)N0c * 4, stream);
  hipMemsetAsync(deg1, 0, (size_t)N1c * 4, stream);
  hipMemsetAsync(deg2, 0, (size_t)N2c * 4, stream);
  deg_kernel<<<(E0c + B - 1) / B, B, 0, stream>>>(ei0, E0c, deg0);
  deg_kernel<<<(E1c + B - 1) / B, B, 0, stream>>>(ei1, E1c, deg1);
  deg_kernel<<<(E2c + B - 1) / B, B, 0, stream>>>(ei2, E2c, deg2);

  // Level 0: conv1a (1->32, K=5), conv1b (32->32, K=5).
  run_conv(x,   w1a, r1a, b1a, eat0, ei0, E0c, deg0, zbuf, agg, h0a, N0c, 1,  32, 5, stream);
  run_conv(h0a, w1b, r1b, b1b, eat0, ei0, E0c, deg0, zbuf, agg, h0b, N0c, 32, 32, 5, stream);

  // pool1: segment_max features, segment_mean positions; ea1 = Cartesian.
  {
    int n = N1c * 32;
    fill_i32_kernel<<<(n + B - 1) / B, B, 0, stream>>>(aggI, (int)0x80000000, n);
    pool_max_kernel<<<(N0c * 32 + B - 1) / B, B, 0, stream>>>(h0b, cl1, aggI, N0c, 32);
    unmap_kernel<<<(n + B - 1) / B, B, 0, stream>>>(aggI, x1pool, n);
    hipMemsetAsync(psum, 0, (size_t)N1c * 2 * 4, stream);
    hipMemsetAsync(pcnt, 0, (size_t)N1c * 4, stream);
    pos_accum_kernel<<<(N0c + B - 1) / B, B, 0, stream>>>(pos, cl1, psum, pcnt, N0c);
    pos_div_kernel<<<(N1c + B - 1) / B, B, 0, stream>>>(psum, pcnt, pos1, N1c);
    hipMemsetAsync(maxI, 0, 4, stream);
    cart_max_kernel<<<(E1c + B - 1) / B, B, 0, stream>>>(pos1, ei1, E1c, maxI);
    cart_ea_kernel<<<(E1c + B - 1) / B, B, 0, stream>>>(pos1, ei1, E1c, maxI, ea1);
  }

  // Level 1: conv2a (32->64, K=3), conv2b (64->64, K=3).
  run_conv(x1pool, w2a, r2a, b2a, ea1, ei1, E1c, deg1, zbuf, agg, x1a, N1c, 32, 64, 3, stream);
  run_conv(x1a,    w2b, r2b, b2b, ea1, ei1, E1c, deg1, zbuf, agg, x1b, N1c, 64, 64, 3, stream);

  // pool2 + ea2.
  {
    int n = N2c * 64;
    fill_i32_kernel<<<(n + B - 1) / B, B, 0, stream>>>(aggI, (int)0x80000000, n);
    pool_max_kernel<<<(N1c * 64 + B - 1) / B, B, 0, stream>>>(x1b, cl2, aggI, N1c, 64);
    unmap_kernel<<<(n + B - 1) / B, B, 0, stream>>>(aggI, x2pool, n);
    hipMemsetAsync(psum, 0, (size_t)N2c * 2 * 4, stream);
    hipMemsetAsync(pcnt, 0, (size_t)N2c * 4, stream);
    pos_accum_kernel<<<(N1c + B - 1) / B, B, 0, stream>>>(pos1, cl2, psum, pcnt, N1c);
    pos_div_kernel<<<(N2c + B - 1) / B, B, 0, stream>>>(psum, pcnt, pos2, N2c);
    hipMemsetAsync(maxI, 0, 4, stream);
    cart_max_kernel<<<(E2c + B - 1) / B, B, 0, stream>>>(pos2, ei2, E2c, maxI);
    cart_ea_kernel<<<(E2c + B - 1) / B, B, 0, stream>>>(pos2, ei2, E2c, maxI, ea2);
  }

  // Level 2: conv3a, conv3b (64->64, K=3), convfr (64->1, K=3).
  run_conv(x2pool, w3a, r3a, b3a, ea2, ei2, E2c, deg2, zbuf, agg, x2a, N2c, 64, 64, 3, stream);
  run_conv(x2a,    w3b, r3b, b3b, ea2, ei2, E2c, deg2, zbuf, agg, x2b, N2c, 64, 64, 3, stream);
  run_conv(x2b,    wfr, rfr, bfr, ea2, ei2, E2c, deg2, zbuf, agg, xfr, N2c, 64, 1,  3, stream);

  // Skip branch at level 1: convp1 (32->1, K=3) on pool1_x.
  run_conv(x1pool, wp1, rp1, bp1, ea1, ei1, E1c, deg1, zbuf, agg, sp1, N1c, 32, 1, 3, stream);

  // y1 = xfr[cluster2] + sp1 ; out = sigmoid(y1[cluster1]).
  add_up_kernel<<<(N1c + B - 1) / B, B, 0, stream>>>(xfr, cl2, sp1, y1, N1c);
  sig_out_kernel<<<(N0c + B - 1) / B, B, 0, stream>>>(y1, cl1, (float*)d_out, N0c);
}
